// SupervisedMultiModalContrastiveLoss_81458349736414
// MI455X (gfx1250) — matrix operand order repaired and checked by reading, not compiled, not run on hardware
//
#include <hip/hip_runtime.h>

// ---------------------------------------------------------------------------
// Problem constants (fixed by the reference's setup_inputs)
// ---------------------------------------------------------------------------
#define NUM_CLASSES 19
#define NBATCH      2
#define VIEWS       50
#define NSAMP       1900          // NBATCH * NUM_CLASSES * VIEWS
#define NPAD        1920          // padded to a multiple of 64 for WMMA strips
#define CDIM        256           // feature channels (K of the Gram GEMM)
#define FH          128
#define FW          256
#define HW          (FH * FW)     // 32768 feature pixels per batch
#define LH          512
#define LW          1024
#define INV_TEMP    10.0f         // 1 / 0.1  (TEMP == CM_TEMP)

typedef __attribute__((ext_vector_type(16))) _Float16 v16h;
typedef __attribute__((ext_vector_type(8)))  _Float16 v8h;
typedef __attribute__((ext_vector_type(8)))  float    v8f;

// ---------------------------------------------------------------------------
// 0) zero the three output scalars (harness poisons d_out)
// ---------------------------------------------------------------------------
__global__ void zero_out_kernel(float* __restrict__ out) {
    if (threadIdx.x < 3) out[threadIdx.x] = 0.0f;
}

// ---------------------------------------------------------------------------
// 1) Per-(batch,class) pixel selection. One wave32 per (n,class).
//    Deterministic ballot/popcount stream compaction over the 4x-strided
//    downsampled label: pick the first VIEWS pixels of that class.
// ---------------------------------------------------------------------------
__global__ void build_idx_kernel(const int* __restrict__ label,
                                 int* __restrict__ idxbuf) {
    const int b    = blockIdx.x;               // 0 .. NBATCH*NUM_CLASSES-1
    const int n    = b / NUM_CLASSES;
    const int cls  = b % NUM_CLASSES;
    const int lane = threadIdx.x;              // 0..31 (wave32)
    const int lbase = n * LH * LW;

    int count = 0;
    for (int p0 = 0; p0 < HW && count < VIEWS; p0 += 32) {
        const int p = p0 + lane;
        const int r = p >> 8;                  // row in 128x256 grid
        const int c = p & 255;
        const int lv = label[lbase + (r << 2) * LW + (c << 2)];  // ::4,::4
        const bool match = (lv == cls);
        const unsigned long long m = __ballot(match);            // low 32 bits
        const int rank = __popcll(m & ((1ull << lane) - 1ull));
        if (match) {
            const int slot = count + rank;
            if (slot < VIEWS) idxbuf[b * VIEWS + slot] = p;
        }
        count += __popcll(m);
    }
    // degenerate class with < VIEWS pixels: pad deterministically with pixel 0
    for (int s = count + lane; s < VIEWS; s += 32) idxbuf[b * VIEWS + s] = 0;
}

// ---------------------------------------------------------------------------
// 2) Gather + L2-normalize both modalities. One wave32 per sample row.
//    Output: f16 matrices [NPAD, CDIM], rows >= NSAMP zeroed.
// ---------------------------------------------------------------------------
__global__ void gather_norm_kernel(const float* __restrict__ cf,
                                   const float* __restrict__ of,
                                   const int* __restrict__ idxbuf,
                                   _Float16* __restrict__ fcf,
                                   _Float16* __restrict__ fof) {
    const int wave = threadIdx.x >> 5;         // 8 waves / block
    const int lane = threadIdx.x & 31;
    const int s = blockIdx.x * 8 + wave;       // sample row
    if (s >= NPAD) return;

    if (s >= NSAMP) {                          // zero padding rows
        for (int j = 0; j < 8; ++j) {
            const int c = lane + 32 * j;
            fcf[s * CDIM + c] = (_Float16)0.0f;
            fof[s * CDIM + c] = (_Float16)0.0f;
        }
        return;
    }

    const int t   = s / VIEWS;                 // 0..37  (n*19 + class)
    const int v   = s % VIEWS;
    const int n   = t / NUM_CLASSES;
    const int pix = idxbuf[t * VIEWS + v];
    const long base = (long)n * CDIM * HW + pix;

    float vc[8], vo[8];
    float ssc = 0.0f, sso = 0.0f;
#pragma unroll
    for (int j = 0; j < 8; ++j) {
        const int c = lane + 32 * j;
        const float a  = cf[base + (long)c * HW];
        const float b2 = of[base + (long)c * HW];
        vc[j] = a;  vo[j] = b2;
        ssc += a * a;  sso += b2 * b2;
    }
#pragma unroll
    for (int m = 16; m >= 1; m >>= 1) {        // wave32 reduction
        ssc += __shfl_xor(ssc, m, 32);
        sso += __shfl_xor(sso, m, 32);
    }
    const float sc = 1.0f / fmaxf(sqrtf(ssc), 1e-12f);
    const float so = 1.0f / fmaxf(sqrtf(sso), 1e-12f);
#pragma unroll
    for (int j = 0; j < 8; ++j) {
        const int c = lane + 32 * j;
        fcf[s * CDIM + c] = (_Float16)(vc[j] * sc);
        fof[s * CDIM + c] = (_Float16)(vo[j] * so);
    }
}

// ---------------------------------------------------------------------------
// 3) Gram GEMM: D[i][j] = sum_k A[i][k] * B[j][k]   (A,B: [NPAD,CDIM] f16)
//    One wave32 per 16x64 output strip: 4 accumulators, 8 K-steps, 32 WMMAs.
//    The A fragment is loaded once per K-step and reused by 4 WMMAs,
//    cutting the load:wmma ratio from 4:1 to 2.5:1.
//
//    Layouts used here (wave32; contiguous k for A and B, not the ISA table order), lo = lane&15, hi = lane>>4:
//      A (16x32 f16): M = lo; half h -> K = hi*16 + h
//                     => two contiguous 8-half (16B) loads at kk+hi*16, kk+hi*16+8
//      B (32x16 f16): N = lo; half h -> K = hi*16 + h
//                     => one contiguous 16-half (32B) load at kk+hi*16
//      C/D (16x16 f32): VGPR r -> M = hi*8 + r, N = lo
// ---------------------------------------------------------------------------
__global__ void gram_wmma_kernel(const _Float16* __restrict__ A,
                                 const _Float16* __restrict__ B,
                                 float* __restrict__ D) {
    const int tm   = blockIdx.x;               // 0..NPAD/16-1   (M tile)
    const int tn4  = blockIdx.y;               // 0..NPAD/64-1   (N quad-tile)
    const int lane = threadIdx.x;              // 0..31, EXEC all-ones
    const int lo   = lane & 15;
    const int hi   = lane >> 4;

    const _Float16* arow = A + (tm * 16 + lo) * CDIM;
    const _Float16* b0r  = B + (tn4 * 64 +  0 + lo) * CDIM;
    const _Float16* b1r  = B + (tn4 * 64 + 16 + lo) * CDIM;
    const _Float16* b2r  = B + (tn4 * 64 + 32 + lo) * CDIM;
    const _Float16* b3r  = B + (tn4 * 64 + 48 + lo) * CDIM;

    v8f acc0 = {}, acc1 = {}, acc2 = {}, acc3 = {};
#pragma unroll
    for (int kk = 0; kk < CDIM; kk += 32) {
        const v8h a0 = *(const v8h*)(arow + kk + hi * 16);       // 16B aligned
        const v8h a1 = *(const v8h*)(arow + kk + hi * 16 + 8);   // 16B aligned
        v16h a;
#pragma unroll
        for (int e = 0; e < 8; ++e) { a[e] = a0[e]; a[8 + e] = a1[e]; }

        const v16h b0 = *(const v16h*)(b0r + kk + hi * 16);      // 32B aligned
        const v16h b1 = *(const v16h*)(b1r + kk + hi * 16);
        const v16h b2 = *(const v16h*)(b2r + kk + hi * 16);
        const v16h b3 = *(const v16h*)(b3r + kk + hi * 16);

        // (neg_a, A, neg_b, B, c_mod, C, reuse_a, reuse_b)
        acc0 = __builtin_amdgcn_wmma_f32_16x16x32_f16(
                   false, a, false, b0, (short)0, acc0, false, false);
        acc1 = __builtin_amdgcn_wmma_f32_16x16x32_f16(
                   false, a, false, b1, (short)0, acc1, false, false);
        acc2 = __builtin_amdgcn_wmma_f32_16x16x32_f16(
                   false, a, false, b2, (short)0, acc2, false, false);
        acc3 = __builtin_amdgcn_wmma_f32_16x16x32_f16(
                   false, a, false, b3, (short)0, acc3, false, false);
    }

    float* drow = D + (long)(tm * 16 + hi * 8) * NPAD + tn4 * 64 + lo;
#pragma unroll
    for (int r = 0; r < 8; ++r) {
        drow[(long)r * NPAD +  0] = acc0[r];
        drow[(long)r * NPAD + 16] = acc1[r];
        drow[(long)r * NPAD + 32] = acc2[r];
        drow[(long)r * NPAD + 48] = acc3[r];
    }
}

// ---------------------------------------------------------------------------
// 4) InfoNCE row reduction. One 256-thread block per row i (< NSAMP).
//    Pass 1: cache scaled row in LDS, reduce neg exp-sum.
//    Pass 2: accumulate pos log-prob terms.  Row loss atomically averaged.
//    labels_flat[i] = (i / VIEWS) % NUM_CLASSES; each class has
//    NBATCH*VIEWS = 100 members => pos count per row is 99.
// ---------------------------------------------------------------------------
__global__ void infonce_kernel(const float* __restrict__ D,
                               float* __restrict__ out, int oidx) {
    __shared__ float row[NSAMP];
    __shared__ float red[256];
    const int i   = blockIdx.x;                // 0..NSAMP-1
    const int tid = threadIdx.x;
    const int ci  = (i / VIEWS) % NUM_CLASSES;

    float nsum = 0.0f;
    for (int j = tid; j < NSAMP; j += 256) {
        const float d = D[(long)i * NPAD + j] * INV_TEMP;
        row[j] = d;
        const int cj = (j / VIEWS) % NUM_CLASSES;
        if (cj != ci) nsum += __expf(d);
    }
    red[tid] = nsum;
    __syncthreads();
    for (int s2 = 128; s2 > 0; s2 >>= 1) {
        if (tid < s2) red[tid] += red[tid + s2];
        __syncthreads();
    }
    const float neg_sum = red[0];
    __syncthreads();

    float psum = 0.0f;
    for (int j = tid; j < NSAMP; j += 256) {
        const int cj = (j / VIEWS) % NUM_CLASSES;
        if (cj == ci && j != i) {
            const float d = row[j];
            psum += d - __logf(__expf(d) + neg_sum);
        }
    }
    red[tid] = psum;
    __syncthreads();
    for (int s2 = 128; s2 > 0; s2 >>= 1) {
        if (tid < s2) red[tid] += red[tid + s2];
        __syncthreads();
    }
    if (tid == 0) {
        const float pos_cnt = (float)(NBATCH * VIEWS - 1);   // 99
        atomicAdd(&out[oidx], -red[0] / (pos_cnt * (float)NSAMP));
    }
}

// ---------------------------------------------------------------------------
// Launch: label(int32) [2,512,1024], color_fea(f32) [2,256,128,256],
//         other_fea(f32) [2,256,128,256] -> d_out: 3 f32 losses.
// Workspace layout (bytes):
//   [0, 8192)                         : idxbuf  int[38*50]
//   [8192, +NPAD*CDIM*2)              : fcf     f16[1920*256]   (0.94 MB)
//   [.., +NPAD*CDIM*2)                : fof     f16[1920*256]   (0.94 MB)
//   [.., +NPAD*NPAD*4)                : D       f32[1920*1920]  (14.7 MB, reused 3x)
// ---------------------------------------------------------------------------
extern "C" void kernel_launch(void* const* d_in, const int* in_sizes, int n_in,
                              void* d_out, int out_size, void* d_ws, size_t ws_size,
                              hipStream_t stream) {
    const int*   label = (const int*)d_in[0];
    const float* cf    = (const float*)d_in[1];
    const float* of    = (const float*)d_in[2];
    float*       out   = (float*)d_out;

    char* ws = (char*)d_ws;
    int*      idxbuf = (int*)ws;
    _Float16* fcf    = (_Float16*)(ws + 8192);
    _Float16* fof    = (_Float16*)(ws + 8192 + (size_t)NPAD * CDIM * 2);
    float*    Dm     = (float*)   (ws + 8192 + 2 * (size_t)NPAD * CDIM * 2);

    zero_out_kernel<<<1, 32, 0, stream>>>(out);
    build_idx_kernel<<<NBATCH * NUM_CLASSES, 32, 0, stream>>>(label, idxbuf);
    gather_norm_kernel<<<NPAD / 8, 256, 0, stream>>>(cf, of, idxbuf, fcf, fof);

    dim3 gtiles(NPAD / 16, NPAD / 64);   // 120 x 30 strips, 1 wave each

    // loss_cm = InfoNCE(fcf @ fof^T / 0.1)
    gram_wmma_kernel<<<gtiles, 32, 0, stream>>>(fcf, fof, Dm);
    infonce_kernel<<<NSAMP, 256, 0, stream>>>(Dm, out, 0);

    // loss_vis = InfoNCE(fcf @ fcf^T / 0.1)
    gram_wmma_kernel<<<gtiles, 32, 0, stream>>>(fcf, fcf, Dm);
    infonce_kernel<<<NSAMP, 256, 0, stream>>>(Dm, out, 1);

    // loss_aux = InfoNCE(fof @ fof^T / 0.1)
    gram_wmma_kernel<<<gtiles, 32, 0, stream>>>(fof, fof, Dm);
    infonce_kernel<<<NSAMP, 256, 0, stream>>>(Dm, out, 2);
}
